// MultiHeadAttention_16939351016165
// MI455X (gfx1250) — compile-verified, matrix operand order checked by reading, not run on hardware
//
#include <hip/hip_runtime.h>
#include <hip/hip_bf16.h>

// ---------------------------------------------------------------------------
// MI455X (gfx1250) multi-head attention, bf16 WMMA path, fp32 accumulation.
//   H=16, d_model=1024, d_k=64, nb=4, nw=2048  ->  M = nb*nw = 8192 rows.
// GEMMs stage block-shared weight tiles into LDS via the Tensor Data Mover
// (double-buffered, TENSORcnt-synced); attention is a per-wave flash kernel.
// ---------------------------------------------------------------------------

#define NB     4
#define NW     2048
#define DM     1024
#define HEADS  16
#define DK     64
#define MROWS  (NB * NW)           // 8192
#define NEGBIG (-1e9f)

typedef __attribute__((ext_vector_type(16))) __bf16 v16bf;
typedef __attribute__((ext_vector_type(8)))  __bf16 v8bf;
typedef __attribute__((ext_vector_type(8)))  float  v8f;
typedef __attribute__((ext_vector_type(4)))  unsigned int u32x4;
typedef __attribute__((ext_vector_type(4)))  int    i32x4;
typedef __attribute__((ext_vector_type(8)))  int    i32x8;

#define WMMA_BF16(A, B, C) \
    __builtin_amdgcn_wmma_f32_16x16x32_bf16(false, (A), false, (B), (short)0, (C), false, false)

#if __has_builtin(__builtin_amdgcn_tensor_load_to_lds)
#define HAVE_TDM 1
#else
#define HAVE_TDM 0
#endif

#if HAVE_TDM
// ---------------------------------------------------------------------------
// TDM: async-DMA a 2-D bf16 tile (tileK contiguous elems x tileRows rows,
// row stride = strideElems) from global into LDS at byte offset ldsOff.
// D# packing per CDNA5 ISA ch.8 (group0: count/lds/global/type; group1:
// data_size, tensor dims, tile dims, dim0 stride). Groups 2/3 zero (2-D).
// ---------------------------------------------------------------------------
__device__ __forceinline__
void tdm_load_tile_bf16(const void* gptr, unsigned ldsOff,
                        unsigned tileK, unsigned tileRows,
                        unsigned strideElems, unsigned tdim0, unsigned tdim1) {
    unsigned long long ga = (unsigned long long)gptr;
    u32x4 g0;
    g0[0] = 1u;                                             // count=1 (valid user D#)
    g0[1] = ldsOff;                                         // lds_addr (bytes)
    g0[2] = (unsigned)(ga & 0xFFFFFFFFu);                   // global_addr[31:0]
    g0[3] = (unsigned)((ga >> 32) & 0x01FFFFFFu)            // global_addr[56:32]
          | (2u << 30);                                     // type=2 ("image")
    i32x8 g1;
    g1[0] = (int)(1u << 16);                                // data_size=1 -> 2 bytes
    g1[1] = (int)((tdim0 & 0xFFFFu) << 16);                 // tensor_dim0[15:0]
    g1[2] = (int)(((tdim0 >> 16) & 0xFFFFu) | ((tdim1 & 0xFFFFu) << 16));
    g1[3] = (int)(((tdim1 >> 16) & 0xFFFFu) | ((tileK & 0xFFFFu) << 16)); // tile_dim0
    g1[4] = (int)(tileRows & 0xFFFFu);                      // tile_dim1 (tile_dim2=0)
    g1[5] = (int)strideElems;                               // tensor_dim0_stride[31:0]
    g1[6] = 0;                                              // stride hi / dim1_stride lo
    g1[7] = 0;
    i32x4 z4 = {0, 0, 0, 0};
#if defined(__clang_major__) && (__clang_major__ >= 23)
    i32x8 z8 = {0, 0, 0, 0, 0, 0, 0, 0};
    __builtin_amdgcn_tensor_load_to_lds(g0, g1, z4, z4, z8, 0);
#else
    __builtin_amdgcn_tensor_load_to_lds(g0, g1, z4, z4, 0);
#endif
}
#endif // HAVE_TDM

// ---------------------------------------------------------------------------
// fp32 -> bf16 conversion (one pass, bandwidth bound, negligible cost)
// ---------------------------------------------------------------------------
__global__ __launch_bounds__(256)
void cvt_f32_bf16(const float* __restrict__ in, __bf16* __restrict__ out, int n) {
    int i = blockIdx.x * 256 + threadIdx.x;
    if (i < n) out[i] = (__bf16)in[i];
}

// ---------------------------------------------------------------------------
// GEMM:  Y[M x 1024] = X[M x 1024] @ W^T, W stored [1024(out) x 1024(in)].
// Block = 8 waves = 128x64 output tile; the 64x32 weight tile for each K-step
// is shared by all 8 waves, so it is staged once into LDS by the TDM
// (double-buffered: issue K+1 DMA, s_wait_tensorcnt<=1 for K, barrier, MAC).
// A fragments are register double-buffered from global (contiguous 32 B/lane).
// ---------------------------------------------------------------------------
__global__ __launch_bounds__(256)
void gemm_bf16_xwt(const __bf16* __restrict__ X, const __bf16* __restrict__ W,
                   void* __restrict__ Y, int storeF32) {
    __shared__ __attribute__((aligned(64))) __bf16 ldsW[2][64 * 32];  // 2 x 4 KB

    const int lane = threadIdx.x & 31;
    const int wave = threadIdx.x >> 5;
    const int lo = lane & 15, hi = lane >> 4;
    const int m0 = blockIdx.y * 128 + wave * 16;
    const int n0 = blockIdx.x * 64;

    v8f acc[4] = {};
    const __bf16* xrow = X + (size_t)(m0 + lo) * DM;
    const __bf16* wtile = W + (size_t)n0 * DM;

#if HAVE_TDM
    const unsigned ldsOff0 = (unsigned)(unsigned long long)(const void*)&ldsW[0][0];
    const unsigned ldsOff1 = (unsigned)(unsigned long long)(const void*)&ldsW[1][0];
    if (wave == 0)   // one DMA per block; TDM ignores EXEC, issued once per wave
        tdm_load_tile_bf16(wtile, ldsOff0, 32, 64, DM, DM, DM);
#endif

    v16bf a_cur = *(const v16bf*)(xrow + hi * 16);

    for (int k0 = 0; k0 < DM; k0 += 32) {
        const int cur = (k0 >> 5) & 1;
#if HAVE_TDM
        if (wave == 0) {
            if (k0 + 32 < DM) {
                tdm_load_tile_bf16(wtile + (k0 + 32), cur ? ldsOff0 : ldsOff1,
                                   32, 64, DM, DM, DM);
                __builtin_amdgcn_s_wait_tensorcnt((short)1);  // K-step tile landed
            } else {
                __builtin_amdgcn_s_wait_tensorcnt((short)0);
            }
        }
        __syncthreads();                    // tile visible to all 8 waves
        const __bf16* Wt = ldsW[cur];
#else
        // cooperative fallback staging: 256 threads x 8 bf16 = 64x32 tile
        {
            int r = threadIdx.x >> 2, c = (threadIdx.x & 3) * 8;
            *(v8bf*)&ldsW[0][r * 32 + c] =
                *(const v8bf*)(wtile + (size_t)r * DM + k0 + c);
        }
        __syncthreads();
        const __bf16* Wt = ldsW[0];
#endif
        // register double-buffer the A fragment
        v16bf a_next = a_cur;
        if (k0 + 32 < DM) {
            a_next = *(const v16bf*)(xrow + (k0 + 32) + hi * 16);
            __builtin_prefetch(xrow + k0 + 64 + hi * 16, 0, 3);
        }
#pragma unroll
        for (int n = 0; n < 4; ++n) {
            v16bf b = *(const v16bf*)(Wt + (n * 16 + lo) * 32 + hi * 16);
            acc[n] = WMMA_BF16(a_cur, b, acc[n]);
        }
        a_cur = a_next;
        __syncthreads();                    // reads done before buffer reuse
    }

    if (storeF32) {
        float* o = (float*)Y;
#pragma unroll
        for (int n = 0; n < 4; ++n)
#pragma unroll
            for (int r = 0; r < 8; ++r)
                o[(size_t)(m0 + r + 8 * hi) * DM + n0 + n * 16 + lo] = acc[n][r];
    } else {
        __bf16* o = (__bf16*)Y;
#pragma unroll
        for (int n = 0; n < 4; ++n)
#pragma unroll
            for (int r = 0; r < 8; ++r)
                o[(size_t)(m0 + r + 8 * hi) * DM + n0 + n * 16 + lo] = (__bf16)acc[n][r];
    }
}

// ---------------------------------------------------------------------------
// Flash attention. One wave per 16-row q tile; online softmax over 32-key
// tiles. q/k/v projected buffers are [b][i][h*64+d] so per-head slices are
// strided views (no physical head transpose needed).
//   scores: S(16x32) = q(16x64) @ k^T  -> 2 N-tiles x 2 K-steps = 4 WMMA
//   PV:     O(16x64) += P(16x32) @ V(32x64) -> 4 WMMA (V transposed in LDS)
// ---------------------------------------------------------------------------
__global__ __launch_bounds__(256)
void attn_flash(const __bf16* __restrict__ qp, const __bf16* __restrict__ kp,
                const __bf16* __restrict__ vp, const int* __restrict__ mask,
                __bf16* __restrict__ out) {
    __shared__ __attribute__((aligned(32))) __bf16 ldsP[8][16 * 32];   //  8 KB
    __shared__ __attribute__((aligned(32))) __bf16 ldsVT[8][64 * 32];  // 32 KB

    const int lane = threadIdx.x & 31;
    const int wave = threadIdx.x >> 5;
    const int lo = lane & 15, hi = lane >> 4;
    const int qt = blockIdx.x * 8 + wave;    // q tile 0..127
    const int h  = blockIdx.y;
    const int b  = blockIdx.z;
    const int i0 = qt * 16;

    const size_t bOff = (size_t)b * NW * DM;
    const __bf16* qB = qp + bOff + (size_t)h * DK;
    const __bf16* kB = kp + bOff + (size_t)h * DK;
    const __bf16* vB = vp + bOff + (size_t)h * DK;
    const int*    mB = mask + (size_t)b * NW * NW;

    // q A-fragments for this tile, resident for whole key loop (d 0..31, 32..63)
    v16bf aq0 = *(const v16bf*)(qB + (size_t)(i0 + lo) * DM + hi * 16);
    v16bf aq1 = *(const v16bf*)(qB + (size_t)(i0 + lo) * DM + 32 + hi * 16);

    float m[8], l[8];
    v8f acc[4] = {};
#pragma unroll
    for (int r = 0; r < 8; ++r) { m[r] = -3.0e38f; l[r] = 0.0f; }

    __bf16* P  = ldsP[wave];
    __bf16* VT = ldsVT[wave];

    for (int j0 = 0; j0 < NW; j0 += 32) {
        // ---- scores: two 16x16 C tiles, K accumulated over d=0..63 ----
        v16bf bk0a = *(const v16bf*)(kB + (size_t)(j0 + lo) * DM + hi * 16);
        v16bf bk0b = *(const v16bf*)(kB + (size_t)(j0 + lo) * DM + 32 + hi * 16);
        v16bf bk1a = *(const v16bf*)(kB + (size_t)(j0 + 16 + lo) * DM + hi * 16);
        v16bf bk1b = *(const v16bf*)(kB + (size_t)(j0 + 16 + lo) * DM + 32 + hi * 16);
        v8f s0 = {}, s1 = {};
        s0 = WMMA_BF16(aq0, bk0a, s0);
        s0 = WMMA_BF16(aq1, bk0b, s0);
        s1 = WMMA_BF16(aq0, bk1a, s1);
        s1 = WMMA_BF16(aq1, bk1b, s1);

        // ---- scale + mask + row max (C layout: row = r+8*hi, col = lo) ----
        float rmax[8];
#pragma unroll
        for (int r = 0; r < 8; ++r) {
            const int* mrow = mB + (size_t)(i0 + r + 8 * hi) * NW + j0;
            float x0 = s0[r] * 0.125f;              // 1/sqrt(d_k)
            float x1 = s1[r] * 0.125f;
            if (mrow[lo] == 0)      x0 = NEGBIG;
            if (mrow[16 + lo] == 0) x1 = NEGBIG;
            s0[r] = x0; s1[r] = x1;
            rmax[r] = fmaxf(x0, x1);
        }
#pragma unroll
        for (int off = 1; off < 16; off <<= 1)
#pragma unroll
            for (int r = 0; r < 8; ++r)
                rmax[r] = fmaxf(rmax[r], __shfl_xor(rmax[r], off, 16));

        // ---- online softmax update ----
        float rsum[8];
#pragma unroll
        for (int r = 0; r < 8; ++r) {
            float nm = fmaxf(m[r], rmax[r]);
            float corr = __expf(m[r] - nm);
            m[r] = nm;
            float p0 = __expf(s0[r] - nm);
            float p1 = __expf(s1[r] - nm);
            s0[r] = p0; s1[r] = p1;
            rsum[r] = p0 + p1;
            l[r] *= corr;
#pragma unroll
            for (int n = 0; n < 4; ++n) acc[n][r] *= corr;
        }
#pragma unroll
        for (int off = 1; off < 16; off <<= 1)
#pragma unroll
            for (int r = 0; r < 8; ++r)
                rsum[r] += __shfl_xor(rsum[r], off, 16);
#pragma unroll
        for (int r = 0; r < 8; ++r) l[r] += rsum[r];

        // ---- P: C layout -> bf16 A layout via per-wave LDS ----
#pragma unroll
        for (int r = 0; r < 8; ++r) {
            int row = r + 8 * hi;
            P[row * 32 + lo]      = (__bf16)s0[r];
            P[row * 32 + 16 + lo] = (__bf16)s1[r];
        }
        // ---- V tile (32 x 64) -> LDS transposed VT[d][j] ----
#pragma unroll 4
        for (int t = 0; t < 64; ++t) {
            int idx = t * 32 + lane;
            int j = idx >> 6, d = idx & 63;
            VT[d * 32 + j] = vB[(size_t)(j0 + j) * DM + d];
        }
        asm volatile("s_wait_dscnt 0" ::: "memory");   // per-wave LDS RAW fence

        v16bf ap = *(const v16bf*)(P + lo * 32 + hi * 16);
#pragma unroll
        for (int n = 0; n < 4; ++n) {
            v16bf bv = *(const v16bf*)(VT + (size_t)(n * 16 + lo) * 32 + hi * 16);
            acc[n] = WMMA_BF16(ap, bv, acc[n]);
        }
        asm volatile("s_wait_dscnt 0" ::: "memory");   // LDS reads done before overwrite
    }

    // ---- normalize and emit bf16 in [b][i][h*64+d] layout ----
#pragma unroll
    for (int n = 0; n < 4; ++n) {
#pragma unroll
        for (int r = 0; r < 8; ++r) {
            float v = acc[n][r] / l[r];
            out[bOff + (size_t)(i0 + r + 8 * hi) * DM + h * DK + n * 16 + lo] = (__bf16)v;
        }
    }
}

// ---------------------------------------------------------------------------
// Host-side orchestration (graph-capture safe: only kernel launches on stream)
// ---------------------------------------------------------------------------
extern "C" void kernel_launch(void* const* d_in, const int* in_sizes, int n_in,
                              void* d_out, int out_size, void* d_ws, size_t ws_size,
                              hipStream_t stream) {
    const float* Q  = (const float*)d_in[0];
    const float* K  = (const float*)d_in[1];
    const float* V  = (const float*)d_in[2];
    const int* mask = (const int*)d_in[3];
    const float* Wq = (const float*)d_in[4];
    const float* Wk = (const float*)d_in[5];
    const float* Wv = (const float*)d_in[6];
    const float* Wo = (const float*)d_in[7];
    float* out = (float*)d_out;

    const size_t nBig = (size_t)MROWS * DM;   // 8,388,608 elems
    const size_t nW   = (size_t)DM * DM;      // 1,048,576 elems

    // workspace carve-up (bf16 elements)
    __bf16* p = (__bf16*)d_ws;
    __bf16* Qb  = p;  p += nBig;
    __bf16* Kb  = p;  p += nBig;
    __bf16* Vb  = p;  p += nBig;
    __bf16* Wqb = p;  p += nW;
    __bf16* Wkb = p;  p += nW;
    __bf16* Wvb = p;  p += nW;
    __bf16* Wob = p;  p += nW;
    __bf16* qp  = p;  p += nBig;
    __bf16* kp  = p;  p += nBig;
    __bf16* vp  = p;  p += nBig;
    __bf16* ab  = p;  p += nBig;              // attention output, bf16

    // 1) fp32 -> bf16
    {
        int gb = (int)((nBig + 255) / 256), gw = (int)((nW + 255) / 256);
        cvt_f32_bf16<<<gb, 256, 0, stream>>>(Q,  Qb,  (int)nBig);
        cvt_f32_bf16<<<gb, 256, 0, stream>>>(K,  Kb,  (int)nBig);
        cvt_f32_bf16<<<gb, 256, 0, stream>>>(V,  Vb,  (int)nBig);
        cvt_f32_bf16<<<gw, 256, 0, stream>>>(Wq, Wqb, (int)nW);
        cvt_f32_bf16<<<gw, 256, 0, stream>>>(Wk, Wkb, (int)nW);
        cvt_f32_bf16<<<gw, 256, 0, stream>>>(Wv, Wvb, (int)nW);
        cvt_f32_bf16<<<gw, 256, 0, stream>>>(Wo, Wob, (int)nW);
    }

    // 2) Q/K/V projections (bf16 WMMA, TDM-staged weights, bf16 outputs)
    dim3 ggrid(DM / 64, MROWS / 128);         // (16, 64)
    gemm_bf16_xwt<<<ggrid, 256, 0, stream>>>(Qb, Wqb, qp, 0);
    gemm_bf16_xwt<<<ggrid, 256, 0, stream>>>(Kb, Wkb, kp, 0);
    gemm_bf16_xwt<<<ggrid, 256, 0, stream>>>(Vb, Wvb, vp, 0);

    // 3) masked flash attention, all heads/batches
    dim3 agrid(NW / 16 / 8, HEADS, NB);       // (16, 16, 4)
    attn_flash<<<agrid, 256, 0, stream>>>(qp, kp, vp, mask, ab);

    // 4) output projection, fp32 result straight to d_out
    gemm_bf16_xwt<<<ggrid, 256, 0, stream>>>(ab, Wob, out, 1);
}
